// EGNN_33182917329496
// MI455X (gfx1250) — matrix operand order repaired and checked by reading, not compiled, not run on hardware
//
#include <hip/hip_runtime.h>
#include <hip/hip_bf16.h>
#include <math.h>

// ---------------- problem constants ----------------
#define B_       4
#define N_       2048
#define DIM_     128
#define K_       32
#define EDGE_IN  265      // 2*DIM + 2*FF + 1
#define EIN_PAD  288      // 9 * 32
#define KS1      9        // K-steps for edge GEMM1
#define H1       530      // edge hidden
#define H1_PAD   544      // 34 * 16
#define NT1      34       // N-tiles for edge GEMM1
#define KS2      17       // 544 / 32, K-steps for edge GEMM2
#define MD       16       // m_ij dim
#define NODE_IN  144      // DIM + MD
#define NIN_PAD  160      // 5 * 32
#define KS3      5
#define NH       256      // node hidden
#define NT3      16
#define KS4      8        // 256 / 32
#define NT4      8        // 128 / 16

typedef __attribute__((ext_vector_type(16))) _Float16 v16h;
typedef __attribute__((ext_vector_type(8)))  _Float16 v8h;
typedef __attribute__((ext_vector_type(8)))  float    v8f;

__device__ __forceinline__ float silu_f(float x) { return x / (1.f + __expf(-x)); }

__device__ __forceinline__ v8f wmma_f16(v16h a, v16h b, v8f c) {
  // D = A(16x32 f16) * B(32x16 f16) + C(16x16 f32)
  return __builtin_amdgcn_wmma_f32_16x16x32_f16(false, a, false, b, (short)0, c, false, false);
}

// A-fragment (16x32 f16) from row-major LDS tile.
// Lane l<16 : row M=l,  K = {0..7, 16..23};  lane l>=16: row M=l-16, K = {8..15, 24..31}.
__device__ __forceinline__ v16h load_a_frag(const _Float16* base, int stride, int row, int ks, int kb) {
  union { v16h v; v8h p[2]; } u;
  const _Float16* p = base + row * stride + ks * 32;
  u.p[0] = *(const v8h*)(p + kb);        // 16 bytes
  u.p[1] = *(const v8h*)(p + 16 + kb);   // 16 bytes
  return u.v;
}

// ---------------- kernel 0: weight pre-swizzle to WMMA B layout ----------------
// B fragment (32x16): lane n<16 holds column N=n, K={0..7,16..23}; lane 16+n holds N=n, K={8..15,24..31}.
// Stored so each lane's 16 halves are contiguous (one 32B load).
__device__ __forceinline__ void swz(const float* __restrict__ W, _Float16* __restrict__ dst,
                                    int Kreal, int Nreal, int KS, int NT, int tid, int nth) {
  int total = KS * NT * 512;   // 32 lanes * 16 elems per fragment
  for (int x = tid; x < total; x += nth) {
    int e    = x & 15;
    int lane = (x >> 4) & 31;
    int f    = x >> 9;
    int nt   = f % NT;
    int ks   = f / NT;
    int k = ks * 32 + ((lane >> 4) << 3) + (e & 7) + ((e >> 3) << 4);
    int n = nt * 16 + (lane & 15);
    float v = (k < Kreal && n < Nreal) ? W[(size_t)k * Nreal + n] : 0.f;
    dst[x] = (_Float16)v;
  }
}

__global__ void egnn_prep_kernel(const float* __restrict__ eW1, const float* __restrict__ eW2,
                                 const float* __restrict__ nW1, const float* __restrict__ nW2,
                                 _Float16* __restrict__ eW1sw, _Float16* __restrict__ eW2sw,
                                 _Float16* __restrict__ nW1sw, _Float16* __restrict__ nW2sw) {
  int tid = blockIdx.x * blockDim.x + threadIdx.x;
  int nth = gridDim.x * blockDim.x;
  swz(eW1, eW1sw, EDGE_IN, H1,   KS1, NT1, tid, nth);
  swz(eW2, eW2sw, H1,      MD,   KS2, 1,   tid, nth);
  swz(nW1, nW1sw, NODE_IN, NH,   KS3, NT3, tid, nth);
  swz(nW2, nW2sw, NH,      DIM_, KS4, NT4, tid, nth);
}

// ---------------- kernel 1: brute-force kNN (K smallest sq-dists, incl. self) ----------------
__global__ void egnn_knn_kernel(const float* __restrict__ coors, int* __restrict__ knn) {
  __shared__ float sDist[N_];
  __shared__ float sRv[256];
  __shared__ int   sRi[256];
  int node = blockIdx.x;
  int b    = node / N_;
  int tid  = threadIdx.x;
  float cx = coors[(size_t)node * 3 + 0];
  float cy = coors[(size_t)node * 3 + 1];
  float cz = coors[(size_t)node * 3 + 2];
  for (int j = tid; j < N_; j += 256) {
    const float* cj = coors + ((size_t)b * N_ + j) * 3;
    float dx = cx - cj[0], dy = cy - cj[1], dz = cz - cj[2];
    sDist[j] = dx * dx + dy * dy + dz * dz;
  }
  __syncthreads();
  int* out = knn + (size_t)node * K_;
  for (int k = 0; k < K_; ++k) {
    float bv = 3.0e38f; int bi = N_;
    for (int j = tid; j < N_; j += 256) {
      float v = sDist[j];
      if (v < bv) { bv = v; bi = j; }
    }
    sRv[tid] = bv; sRi[tid] = bi;
    __syncthreads();
    for (int s = 128; s > 0; s >>= 1) {
      if (tid < s) {
        float ov = sRv[tid + s]; int oi = sRi[tid + s];
        if (ov < sRv[tid] || (ov == sRv[tid] && oi < sRi[tid])) { sRv[tid] = ov; sRi[tid] = oi; }
      }
      __syncthreads();
    }
    if (tid == 0) { out[k] = sRi[0]; sDist[sRi[0]] = 3.0e38f; }
    __syncthreads();
  }
}

// ---------------- kernel 2: edge MLP (WMMA) + coor MLP + reductions ----------------
__global__ void __launch_bounds__(256)
egnn_edge_kernel(const float* __restrict__ feats, const float* __restrict__ coors,
                 const int* __restrict__ knn,
                 const _Float16* __restrict__ eW1sw, const float* __restrict__ eb1,
                 const _Float16* __restrict__ eW2sw, const float* __restrict__ eb2,
                 const float* __restrict__ cW1, const float* __restrict__ cb1,
                 const float* __restrict__ cW2, const float* __restrict__ cb2,
                 float* __restrict__ m_i, float* __restrict__ coors_out) {
  __shared__ __align__(32) _Float16 sEdge[K_ * EIN_PAD];   // 18 KB: edge inputs (A of GEMM1)
  __shared__ __align__(32) _Float16 sH[K_ * H1_PAD];       // 34 KB: hidden (A of GEMM2)
  __shared__ float sM[K_ * MD];                            // m_ij (f32)
  __shared__ float sRel[K_ * 3];
  __shared__ float sD[K_];
  __shared__ int   sJ[K_];
  __shared__ float sRed[K_ * 3];

  int node = blockIdx.x;
  int b    = node / N_;
  int tid  = threadIdx.x;

  // ---- phase 1: neighbor gather + fourier encode into sEdge ----
  if (tid < K_) {
    int j = knn[(size_t)node * K_ + tid];
    sJ[tid] = j;
    const float* ci = coors + (size_t)node * 3;
    const float* cj = coors + ((size_t)b * N_ + j) * 3;
    float rx = ci[0] - cj[0], ry = ci[1] - cj[1], rz = ci[2] - cj[2];
    sRel[tid * 3 + 0] = rx; sRel[tid * 3 + 1] = ry; sRel[tid * 3 + 2] = rz;
    sD[tid] = rx * rx + ry * ry + rz * rz;
  }
  __syncthreads();
  {
    int e = tid >> 3;          // edge 0..31
    int part = tid & 7;        // 8 threads per edge, 36 cols each (8*36 = 288)
    int j = sJ[e];
    float d = sD[e];
    const float* fi = feats + (size_t)node * DIM_;
    const float* fj = feats + ((size_t)b * N_ + j) * DIM_;
    for (int c = part * 36; c < part * 36 + 36; ++c) {
      float v;
      if      (c < 128) v = fi[c];
      else if (c < 256) v = fj[c - 128];
      else if (c < 260) v = __sinf(d / (float)(1 << (c - 256)));
      else if (c < 264) v = __cosf(d / (float)(1 << (c - 260)));
      else if (c == 264) v = d;
      else v = 0.f;
      sEdge[e * EIN_PAD + c] = (_Float16)v;
    }
  }
  __syncthreads();

  int wave = tid >> 5, lane = tid & 31;
  int mrow = lane & 15;
  int kb   = (lane >> 4) * 8;

  // ---- phase 2: GEMM1  h = silu(Ein(32x288) @ eW1(288x544) + b1) ----
  for (int nt = wave; nt < NT1; nt += 8) {
    v8f acc0 = {}; v8f acc1 = {};
    for (int ks = 0; ks < KS1; ++ks) {
      const _Float16* bp = eW1sw + ((size_t)(ks * NT1 + nt) * 32 + lane) * 16;
      if (ks + 1 < KS1) __builtin_prefetch(eW1sw + ((size_t)((ks + 1) * NT1 + nt) * 32 + lane) * 16, 0, 3);
      v16h bf = *(const v16h*)bp;
      acc0 = wmma_f16(load_a_frag(sEdge, EIN_PAD, mrow,      ks, kb), bf, acc0);
      acc1 = wmma_f16(load_a_frag(sEdge, EIN_PAD, 16 + mrow, ks, kb), bf, acc1);
    }
    int col = nt * 16 + (lane & 15);
    float bias = (col < H1) ? eb1[col] : 0.f;
    for (int r = 0; r < 8; ++r) {
      int row = (lane >> 4) * 8 + r;          // D layout: VGPR r -> rows r / 8+r
      sH[row * H1_PAD + col]        = (_Float16)silu_f(acc0[r] + bias);
      sH[(16 + row) * H1_PAD + col] = (_Float16)silu_f(acc1[r] + bias);
    }
  }
  __syncthreads();

  // ---- phase 3: GEMM2  m_ij = silu(h(32x544) @ eW2(544x16) + b2), wave 0 only ----
  if (wave == 0) {
    v8f acc0 = {}; v8f acc1 = {};
    for (int ks = 0; ks < KS2; ++ks) {
      v16h bf = *(const v16h*)(eW2sw + ((size_t)ks * 32 + lane) * 16);
      acc0 = wmma_f16(load_a_frag(sH, H1_PAD, mrow,      ks, kb), bf, acc0);
      acc1 = wmma_f16(load_a_frag(sH, H1_PAD, 16 + mrow, ks, kb), bf, acc1);
    }
    int col = lane & 15;
    float bias = eb2[col];
    for (int r = 0; r < 8; ++r) {
      int row = (lane >> 4) * 8 + r;
      sM[row * MD + col]        = silu_f(acc0[r] + bias);
      sM[(16 + row) * MD + col] = silu_f(acc1[r] + bias);
    }
  }
  __syncthreads();

  // ---- phase 4: coor MLP (16->64->1) per edge, reductions ----
  if (tid < K_) {
    int e = tid;
    float m[MD];
    for (int d = 0; d < MD; ++d) m[d] = sM[e * MD + d];
    float cw = cb2[0];
    for (int o = 0; o < 64; ++o) {
      float t = cb1[o];
      for (int d = 0; d < MD; ++d) t += m[d] * cW1[d * 64 + o];
      cw += silu_f(t) * cW2[o];
    }
    sRed[e * 3 + 0] = cw * sRel[e * 3 + 0];
    sRed[e * 3 + 1] = cw * sRel[e * 3 + 1];
    sRed[e * 3 + 2] = cw * sRel[e * 3 + 2];
  }
  if (tid < MD) {                 // m_i = sum_j m_ij
    float s = 0.f;
    for (int e = 0; e < K_; ++e) s += sM[e * MD + tid];
    m_i[(size_t)node * MD + tid] = s;
  }
  __syncthreads();
  if (tid == 0) {
    float sx = 0.f, sy = 0.f, sz = 0.f;
    for (int e = 0; e < K_; ++e) { sx += sRed[e * 3]; sy += sRed[e * 3 + 1]; sz += sRed[e * 3 + 2]; }
    coors_out[(size_t)node * 3 + 0] = sx + coors[(size_t)node * 3 + 0];
    coors_out[(size_t)node * 3 + 1] = sy + coors[(size_t)node * 3 + 1];
    coors_out[(size_t)node * 3 + 2] = sz + coors[(size_t)node * 3 + 2];
  }
}

// ---------------- kernel 3: node MLP (WMMA) 144 -> 256 -> 128 + residual ----------------
__global__ void __launch_bounds__(256)
egnn_node_kernel(const float* __restrict__ feats, const float* __restrict__ m_i,
                 const _Float16* __restrict__ nW1sw, const float* __restrict__ nb1,
                 const _Float16* __restrict__ nW2sw, const float* __restrict__ nb2,
                 float* __restrict__ node_out) {
  __shared__ __align__(32) _Float16 sIn[16 * NIN_PAD];
  __shared__ __align__(32) _Float16 sH2[16 * NH];
  int tid  = threadIdx.x;
  int base = blockIdx.x * 16;   // 16 nodes per tile

  for (int x = tid; x < 16 * NIN_PAD; x += 256) {
    int r = x / NIN_PAD, c = x % NIN_PAD;
    size_t nd = (size_t)(base + r);
    float v = (c < DIM_) ? feats[nd * DIM_ + c]
            : (c < NODE_IN) ? m_i[nd * MD + (c - DIM_)] : 0.f;
    sIn[x] = (_Float16)v;
  }
  __syncthreads();

  int wave = tid >> 5, lane = tid & 31;
  int mrow = lane & 15;
  int kb   = (lane >> 4) * 8;

  for (int nt = wave; nt < NT3; nt += 8) {
    v8f acc = {};
    for (int ks = 0; ks < KS3; ++ks) {
      v16h bf = *(const v16h*)(nW1sw + ((size_t)(ks * NT3 + nt) * 32 + lane) * 16);
      acc = wmma_f16(load_a_frag(sIn, NIN_PAD, mrow, ks, kb), bf, acc);
    }
    int col = nt * 16 + (lane & 15);
    float bias = nb1[col];
    for (int r = 0; r < 8; ++r) {
      int row = (lane >> 4) * 8 + r;
      sH2[row * NH + col] = (_Float16)silu_f(acc[r] + bias);
    }
  }
  __syncthreads();

  {
    int nt = wave;   // 8 output tiles, 8 waves
    v8f acc = {};
    for (int ks = 0; ks < KS4; ++ks) {
      v16h bf = *(const v16h*)(nW2sw + ((size_t)(ks * NT4 + nt) * 32 + lane) * 16);
      acc = wmma_f16(load_a_frag(sH2, NH, mrow, ks, kb), bf, acc);
    }
    int col = nt * 16 + (lane & 15);
    float bias = nb2[col];
    for (int r = 0; r < 8; ++r) {
      int row = (lane >> 4) * 8 + r;
      size_t nd = (size_t)(base + row);
      node_out[nd * DIM_ + col] = acc[r] + bias + feats[nd * DIM_ + col];
    }
  }
}

// ---------------- launcher ----------------
extern "C" void kernel_launch(void* const* d_in, const int* in_sizes, int n_in,
                              void* d_out, int out_size, void* d_ws, size_t ws_size,
                              hipStream_t stream) {
  const float* feats = (const float*)d_in[0];
  const float* coors = (const float*)d_in[1];
  const float* eW1 = (const float*)d_in[2];
  const float* eb1 = (const float*)d_in[3];
  const float* eW2 = (const float*)d_in[4];
  const float* eb2 = (const float*)d_in[5];
  const float* cW1 = (const float*)d_in[6];
  const float* cb1 = (const float*)d_in[7];
  const float* cW2 = (const float*)d_in[8];
  const float* cb2 = (const float*)d_in[9];
  const float* nW1 = (const float*)d_in[10];
  const float* nb1 = (const float*)d_in[11];
  const float* nW2 = (const float*)d_in[12];
  const float* nb2 = (const float*)d_in[13];

  float* out_nodes = (float*)d_out;                                 // (B,N,DIM)
  float* out_coors = out_nodes + (size_t)B_ * N_ * DIM_;            // (B,N,3)

  // workspace partition (all offsets keep >=32B alignment)
  char* ws = (char*)d_ws;
  int*      knn   = (int*)ws;        ws += (size_t)B_ * N_ * K_ * sizeof(int);       // 1 MB
  float*    m_i   = (float*)ws;      ws += (size_t)B_ * N_ * MD * sizeof(float);     // 512 KB
  _Float16* eW1sw = (_Float16*)ws;   ws += (size_t)KS1 * NT1 * 512 * sizeof(_Float16);
  _Float16* eW2sw = (_Float16*)ws;   ws += (size_t)KS2 * 1   * 512 * sizeof(_Float16);
  _Float16* nW1sw = (_Float16*)ws;   ws += (size_t)KS3 * NT3 * 512 * sizeof(_Float16);
  _Float16* nW2sw = (_Float16*)ws;   ws += (size_t)KS4 * NT4 * 512 * sizeof(_Float16);

  egnn_prep_kernel<<<256, 256, 0, stream>>>(eW1, eW2, nW1, nW2, eW1sw, eW2sw, nW1sw, nW2sw);
  egnn_knn_kernel<<<B_ * N_, 256, 0, stream>>>(coors, knn);
  egnn_edge_kernel<<<B_ * N_, 256, 0, stream>>>(feats, coors, knn,
                                                eW1sw, eb1, eW2sw, eb2,
                                                cW1, cb1, cW2, cb2,
                                                m_i, out_coors);
  egnn_node_kernel<<<(B_ * N_) / 16, 256, 0, stream>>>(feats, m_i, nW1sw, nb1, nW2sw, nb2, out_nodes);
}
